// SeqModel_42417097016494
// MI455X (gfx1250) — matrix operand order repaired and checked by reading, not compiled, not run on hardware
//
#include <hip/hip_runtime.h>

typedef __attribute__((ext_vector_type(16))) _Float16 v16h;
typedef __attribute__((ext_vector_type(8)))  float    v8f;

#define TG        32
#define START_TAG 30
#define STOP_TAG  31
#define LSEQ      512
#define BSZ       256

#define LOG2E 1.4426950408889634f
#define LN2   0.6931471805599453f

union V16H { v16h v; _Float16 h[16]; unsigned u[8]; };

__device__ __forceinline__ unsigned pack_h2(float a, float b) {
  union { _Float16 h[2]; unsigned u; } x;
  x.h[0] = (_Float16)a; x.h[1] = (_Float16)b;
  return x.u;
}

__device__ __forceinline__ float fast_exp(float x) {  // raw v_exp_f32
  return __builtin_amdgcn_exp2f(x * LOG2E);
}
__device__ __forceinline__ float fast_log(float x) {  // raw v_log_f32
  return __builtin_amdgcn_logf(x) * LN2;
}

// One wave owns 16 batches (columns). State in exp-domain: q[tag,b] = exp(p - m_b),
// normalized so max_tag q = 1, plus running offset m_b.
//
// Tag->matrix-row permutation  tag(t, M) = 16*(M/8) + 8t + (M%8)  makes the WMMA
// D output land lane-local in B-fragment order:
//   lane (b,h):  d0[r] -> tag 16h+r,  d1[r] -> tag 16h+8+r   == B slots K=s+16h.
// So the next step's B fragment needs NO cross-lane exchange; the only cross-lane
// op per step is one xor-16 max. A fragments absorb the permutation statically.
__global__ __launch_bounds__(32) void crf_forward_kernel(
    const float* __restrict__ feats, const unsigned char* __restrict__ mask,
    const float* __restrict__ trans, float* __restrict__ ws)
{
  const int lane = threadIdx.x & 31;
  const int col  = lane & 15;   // batch column (B/C/D); row M for A fragment
  const int h    = lane >> 4;
  const int batch = blockIdx.x * 16 + col;

  // A fragments: A_t[M=col][K] = exp(trans[K][ 16*(col/8) + 8t + col%8 ])
  V16H a0, a1;
  {
    const int jb = 16 * (col >> 3) + (col & 7);
#pragma unroll
    for (int s = 0; s < 16; ++s) {
      const int K = s + 16 * h;  // contiguous: K = 16h + s
      a0.h[s] = (_Float16)fast_exp(trans[K * TG + jb]);
      a1.h[s] = (_Float16)fast_exp(trans[K * TG + jb + 8]);
    }
  }

  // State: q0[r] <-> tag 16h+r, q1[r] <-> tag 16h+8+r (lane's 16 contiguous tags)
  float q0[8], q1[8], m;
  {
    const float* fb = feats + ((size_t)batch * LSEQ) * TG + 16 * h;
    float p0[8], p1[8];
    float mx = -3.0e38f;
#pragma unroll
    for (int r = 0; r < 8; ++r) {
      const int j0 = 16 * h + r, j1 = j0 + 8;
      p0[r] = fb[r]     + trans[START_TAG * TG + j0];
      p1[r] = fb[8 + r] + trans[START_TAG * TG + j1];
      mx = fmaxf(mx, fmaxf(p0[r], p1[r]));
    }
    mx = fmaxf(mx, __shfl_xor(mx, 16, 32));
    m = mx;
#pragma unroll
    for (int r = 0; r < 8; ++r) {
      q0[r] = fast_exp(p0[r] - mx);
      q1[r] = fast_exp(p1[r] - mx);
    }
  }

  const unsigned char* mrow = mask + (size_t)batch * LSEQ;

  for (int t = 1; t < LSEQ; ++t) {
    // This lane's 16 contiguous feats values: tags 16h .. 16h+15
    const float4* fb4 =
        reinterpret_cast<const float4*>(feats + ((size_t)batch * LSEQ + t) * TG + 16 * h);
    const float4 fA = fb4[0], fB = fb4[1], fC = fb4[2], fD = fb4[3];
    const float f0[8] = {fA.x, fA.y, fA.z, fA.w, fB.x, fB.y, fB.z, fB.w};
    const float f1[8] = {fC.x, fC.y, fC.z, fC.w, fD.x, fD.y, fD.z, fD.w};

    // B fragment: lane-local, slot s holds q[tag = s + 16h]
    V16H bf;
#pragma unroll
    for (int v = 0; v < 4; ++v) {
      bf.u[v]     = pack_h2(q0[2 * v], q0[2 * v + 1]);
      bf.u[4 + v] = pack_h2(q1[2 * v], q1[2 * v + 1]);
    }

    const v8f c = {};
    v8f d0 = __builtin_amdgcn_wmma_f32_16x16x32_f16(false, a0.v, false, bf.v,
                                                    (short)0, c, false, false);
    v8f d1 = __builtin_amdgcn_wmma_f32_16x16x32_f16(false, a1.v, false, bf.v,
                                                    (short)0, c, false, false);

    // u = S * exp(feats); M = max over all 32 tags (16 local + partner half)
    float u0[8], u1[8];
    float M = 0.f;
#pragma unroll
    for (int r = 0; r < 8; ++r) {
      u0[r] = d0[r] * fast_exp(f0[r]);
      u1[r] = d1[r] * fast_exp(f1[r]);
      M = fmaxf(M, fmaxf(u0[r], u1[r]));
    }
    M = fmaxf(M, __shfl_xor(M, 16, 32));

    const bool mm = mrow[t] != 0;
    const float rM = __builtin_amdgcn_rcpf(M);
    const float mNew = m + fast_log(M);   // one log per step per batch column
#pragma unroll
    for (int r = 0; r < 8; ++r) {
      q0[r] = mm ? (u0[r] * rM) : q0[r];
      q1[r] = mm ? (u1[r] * rM) : q1[r];
    }
    m = mm ? mNew : m;
  }

  // final[b] = m + log( sum_tag q[tag,b] * exp(trans[tag, STOP]) )
  float s = 0.f;
#pragma unroll
  for (int r = 0; r < 8; ++r) {
    const int j0 = 16 * h + r, j1 = j0 + 8;
    s += q0[r] * fast_exp(trans[j0 * TG + STOP_TAG]);
    s += q1[r] * fast_exp(trans[j1 * TG + STOP_TAG]);
  }
  s += __shfl_xor(s, 16, 32);

  float fin = (h == 0) ? (m + fast_log(s)) : 0.f;  // count each batch once
#pragma unroll
  for (int off = 16; off > 0; off >>= 1) fin += __shfl_xor(fin, off, 32);
  if (lane == 0) ws[blockIdx.x] = fin;  // per-tile forward partial
}

// Gold (path) score: one thread per batch, fixed-order block reduction.
__global__ __launch_bounds__(256) void crf_gold_kernel(
    const float* __restrict__ feats, const unsigned char* __restrict__ mask,
    const int* __restrict__ tags, const float* __restrict__ trans,
    float* __restrict__ ws)
{
  const int b = threadIdx.x;
  const unsigned char* mrow = mask + (size_t)b * LSEQ;
  const int* trow = tags + (size_t)b * LSEQ;
  const float* frow = feats + (size_t)b * LSEQ * TG;

  float g = 0.f;
  int prev = START_TAG;
  for (int t = 0; t < LSEQ; ++t) {
    if (mrow[t]) {
      const int cur = trow[t];
      g += frow[t * TG + cur] + trans[prev * TG + cur];
      prev = cur;
    }
  }
  g += trans[prev * TG + STOP_TAG];  // end transition of last valid tag

  __shared__ float red[256];
  red[b] = g;
  __syncthreads();
  for (int off = 128; off > 0; off >>= 1) {
    if (b < off) red[b] += red[b + off];
    __syncthreads();
  }
  if (b == 0) ws[BSZ / 16] = red[0];
}

__global__ void crf_combine_kernel(const float* __restrict__ ws,
                                   float* __restrict__ out) {
  float fwd = 0.f;
  for (int k = 0; k < BSZ / 16; ++k) fwd += ws[k];
  out[0] = fwd - ws[BSZ / 16];
}

extern "C" void kernel_launch(void* const* d_in, const int* in_sizes, int n_in,
                              void* d_out, int out_size, void* d_ws, size_t ws_size,
                              hipStream_t stream) {
  const float*         feats = (const float*)d_in[0];
  const unsigned char* maskp = (const unsigned char*)d_in[1];  // jnp bool = 1 byte
  const int*           tags  = (const int*)d_in[2];
  const float*         trans = (const float*)d_in[3];
  float* out = (float*)d_out;
  float* ws  = (float*)d_ws;  // [0..15] forward partials, [16] gold score

  crf_forward_kernel<<<dim3(BSZ / 16), dim3(32), 0, stream>>>(feats, maskp, trans, ws);
  crf_gold_kernel<<<dim3(1), dim3(256), 0, stream>>>(feats, maskp, tags, trans, ws);
  crf_combine_kernel<<<dim3(1), dim3(1), 0, stream>>>(ws, out);
}
